// PointerGenerator_33629593928179
// MI455X (gfx1250) — matrix operand order repaired and checked by reading, not compiled, not run on hardware
//
#include <hip/hip_runtime.h>
#include <hip/hip_bf16.h>
#include <cstdint>
#include <cstddef>

// Pointer-generator forward for MI455X (gfx1250).
// All GEMMs use V_WMMA_F32_16X16X32_BF16 (batch B=16 == one M tile).
// Vocab GEMM streams voutW via the Tensor Data Mover (TENSORcnt) into a
// double-buffered LDS tile when the toolchain exposes the TDM builtin.
// Requires ws_size >= ~82 MB.

#define B_   16
#define L_   400
#define T_   64
#define V_   50000
#define E_   128
#define H_   256
#define OOV_ 50
#define VEXT_ (V_ + OOV_)
#define G4H_ (4 * H_)   // 1024
#define H2_  (2 * H_)   // 512
#define H3_  (3 * H_)   // 768

typedef __attribute__((ext_vector_type(16))) __bf16 v16bf;
typedef __attribute__((ext_vector_type(8)))  float  v8f;
typedef __attribute__((ext_vector_type(4)))  unsigned int u32x4;
typedef __attribute__((ext_vector_type(8)))  int i32x8;
typedef __attribute__((ext_vector_type(4)))  int i32x4;

// ---- TDM availability (device pass only; host pass parses the fallback) ----
#if defined(__HIP_DEVICE_COMPILE__) && defined(__has_builtin)
#if __has_builtin(__builtin_amdgcn_tensor_load_to_lds) && \
    __has_builtin(__builtin_amdgcn_s_wait_tensorcnt)
#define HAVE_TDM 1
#endif
#endif
#ifndef HAVE_TDM
#define HAVE_TDM 0
#endif

union FragU { v16bf v; uint4 q[2]; };

__device__ __forceinline__ unsigned short f2bf(float f) {
  union { float f; unsigned int u; } x; x.f = f;
  unsigned int u = x.u + 0x7fffu + ((x.u >> 16) & 1u);   // RNE
  return (unsigned short)(u >> 16);
}
__device__ __forceinline__ float bf2f(unsigned short b) {
  union { float f; unsigned int u; } x; x.u = ((unsigned int)b) << 16; return x.f;
}
__device__ __forceinline__ float sigm(float x) { return 1.0f / (1.0f + __expf(-x)); }
__device__ __forceinline__ v8f zero8() { v8f z = {0.f,0.f,0.f,0.f,0.f,0.f,0.f,0.f}; return z; }

// A fragment (16x32 bf16, MxK) from row-major [M][K] (ISA 7.12.2):
// lane<16: M=lane, K = k0+{0..7} and k0+16+{0..7}; lane>=16: +8 on K halves.
__device__ __forceinline__ v16bf load_a(const unsigned short* base, int ld, int k0) {
  int lane = threadIdx.x & 31;
  const unsigned short* r = base + (lane & 15) * ld + k0 + ((lane >> 4) << 3);
  FragU f;
  f.q[0] = *(const uint4*)(r);
  f.q[1] = *(const uint4*)(r + 16);
  return f.v;
}
// B fragment (32x16 bf16, KxN) from row-major weights [N][K]:
// lane n = n0+(lane&15), K = k0+{0..7} and k0+16+{0..7}; lane>=16: +8 on K halves.
__device__ __forceinline__ v16bf load_b(const unsigned short* base, int ld, int n0, int k0) {
  int lane = threadIdx.x & 31;
  const unsigned short* r = base + (size_t)(n0 + (lane & 15)) * ld + k0 + ((lane >> 4) << 3);
  FragU f;
  f.q[0] = *(const uint4*)(r);
  f.q[1] = *(const uint4*)(r + 16);
  return f.v;
}
#define WMMA_BF16(a, b, c) \
  __builtin_amdgcn_wmma_f32_16x16x32_bf16(false, (a), false, (b), (short)0, (c), false, false)

// ----------------------------- TDM tile fetch -------------------------------
#define VCHUNK_ROWS 64
#define VCHUNKS ((V_ / 16 + 3) / 4)                 // 782 chunks of 4 N-tiles
#define VNITER 8
#define VBLOCKS ((VCHUNKS + VNITER - 1) / VNITER)   // 98

// Issue one TDM descriptor: copy VCHUNK_ROWS rows x H_ bf16 (32 KB) of the
// row-major [V,H] weight matrix into LDS at byte offset lds_off (our LDS
// buffer is the kernel's only __shared__ allocation, so it sits at offset 0).
__device__ __forceinline__ void tdm_issue(const void* gaddr, unsigned lds_off) {
#if HAVE_TDM
  unsigned long long ga = (unsigned long long)(uintptr_t)gaddr;
  u32x4 g0 = { 1u,                                            // count=1, user D#
               lds_off,                                       // lds_addr (bytes)
               (unsigned)ga,                                  // global_addr[31:0]
               ((unsigned)((ga >> 32) & 0x1FFFFFFu)) | (2u << 30) };  // [56:32]|type=2
  i32x8 g1 = { (int)(1u << 16),            // workgroup_mask=0, data_size=2B
               (int)((unsigned)H_ << 16),  // tensor_dim0 = 256 (bits [63:48])
               (int)((unsigned)V_ << 16),  // tensor_dim1 lo16  (bits [95:80])
               (int)((unsigned)H_ << 16),  // tile_dim0 = 256   (bits [127:112])
               VCHUNK_ROWS,                // tile_dim1 = 64, tile_dim2 = 0
               H_,                         // tensor_dim0_stride = 256
               0, 0 };                     // dim1_stride unused (2D)
  i32x4 z4 = { 0, 0, 0, 0 };
#if __clang_major__ >= 23
  i32x8 z8 = { 0, 0, 0, 0, 0, 0, 0, 0 };
  __builtin_amdgcn_tensor_load_to_lds(g0, g1, z4, z4, z8, 0);
#else
  __builtin_amdgcn_tensor_load_to_lds(g0, g1, z4, z4, 0);
#endif
#else
  (void)gaddr; (void)lds_off;
#endif
}

// ---------------------------------------------------------------- utilities
__global__ void cast_bf16_kernel(const float* __restrict__ src,
                                 unsigned short* __restrict__ dst, int n) {
  int i = blockIdx.x * blockDim.x + threadIdx.x;
  if (i < n) dst[i] = f2bf(src[i]);
}

__global__ void embed_enc_kernel(const int* __restrict__ enc_input,
                                 const float* __restrict__ emb,
                                 unsigned short* __restrict__ enc_emb_bf) {
  int idx = blockIdx.x * blockDim.x + threadIdx.x;   // [L,B,E]
  if (idx >= L_ * B_ * E_) return;
  int e = idx & (E_ - 1);
  int lb = idx >> 7;
  int l = lb / B_, b = lb - l * B_;
  int tok = enc_input[b * L_ + l];
  enc_emb_bf[idx] = f2bf(emb[(size_t)tok * E_ + e]);
}

__global__ void embed_dec_kernel(const int* __restrict__ dec_input,
                                 const float* __restrict__ emb,
                                 unsigned short* __restrict__ dec_emb_bf,
                                 float* __restrict__ dec_emb_f) {
  int idx = blockIdx.x * blockDim.x + threadIdx.x;   // [T,B,E]
  if (idx >= T_ * B_ * E_) return;
  int e = idx & (E_ - 1);
  int tb = idx >> 7;
  int tt = tb / B_, b = tb - tt * B_;
  int tok = dec_input[b * T_ + tt];
  float v = emb[(size_t)tok * E_ + e];
  dec_emb_bf[idx] = f2bf(v);
  dec_emb_f[idx] = v;
}

// ----------------------------------------------- encoder LSTM (1 WGP, 8 waves)
// Wave w owns gate columns [w*32, w*32+32): holds c in registers for all of L.
__global__ __launch_bounds__(256) void enc_lstm_kernel(
    const unsigned short* __restrict__ xs_bf,  // [L,B,E] bf16
    const unsigned short* __restrict__ Wih,    // [4H,E]  bf16
    const unsigned short* __restrict__ Whh,    // [4H,H]  bf16
    const float* __restrict__ bias,            // [4H]
    float* __restrict__ hs,                    // [L,B,H] f32, original time order
    float* __restrict__ hF, float* __restrict__ cF,  // [B,H]
    int reverse) {
  __shared__ __align__(16) unsigned short sh_h[B_ * H_];   // 8 KB
  const int tid = threadIdx.x, lane = tid & 31, wave = tid >> 5;
  const int ln = lane & 15, mb = (lane >> 4) << 3;
  for (int i = tid; i < B_ * H_; i += 256) sh_h[i] = 0;
  float creg[2][8];
#pragma unroll
  for (int ti = 0; ti < 2; ++ti)
#pragma unroll
    for (int r = 0; r < 8; ++r) creg[ti][r] = 0.f;
  __syncthreads();

  for (int t = 0; t < L_; ++t) {
    int tt = reverse ? (L_ - 1 - t) : t;
    v8f acc[8];
#pragma unroll
    for (int i = 0; i < 8; ++i) acc[i] = zero8();
    const unsigned short* xrow = xs_bf + (size_t)tt * (B_ * E_);
#pragma unroll
    for (int k0 = 0; k0 < E_; k0 += 32) {
      v16bf a = load_a(xrow, E_, k0);
#pragma unroll
      for (int ti = 0; ti < 2; ++ti)
#pragma unroll
        for (int g = 0; g < 4; ++g)
          acc[ti * 4 + g] = WMMA_BF16(a, load_b(Wih, E_, g * H_ + wave * 32 + ti * 16, k0),
                                      acc[ti * 4 + g]);
    }
#pragma unroll 2
    for (int k0 = 0; k0 < H_; k0 += 32) {
      v16bf a = load_a(sh_h, H_, k0);
#pragma unroll
      for (int ti = 0; ti < 2; ++ti)
#pragma unroll
        for (int g = 0; g < 4; ++g)
          acc[ti * 4 + g] = WMMA_BF16(a, load_b(Whh, H_, g * H_ + wave * 32 + ti * 16, k0),
                                      acc[ti * 4 + g]);
    }
    __syncthreads();   // all A-reads of sh_h done
#pragma unroll
    for (int ti = 0; ti < 2; ++ti) {
      int n = wave * 32 + ti * 16 + ln;
      float bi = bias[0 * H_ + n], bf = bias[1 * H_ + n];
      float bg = bias[2 * H_ + n], bo = bias[3 * H_ + n];
#pragma unroll
      for (int r = 0; r < 8; ++r) {
        float ig = sigm(acc[ti * 4 + 0][r] + bi);
        float fg = sigm(acc[ti * 4 + 1][r] + bf);
        float gg = tanhf(acc[ti * 4 + 2][r] + bg);
        float og = sigm(acc[ti * 4 + 3][r] + bo);
        float c = fg * creg[ti][r] + ig * gg;
        float h = og * tanhf(c);
        creg[ti][r] = c;
        int m = mb + r;
        sh_h[m * H_ + n] = f2bf(h);
        hs[(size_t)tt * (B_ * H_) + m * H_ + n] = h;
      }
    }
    __syncthreads();
  }
#pragma unroll
  for (int ti = 0; ti < 2; ++ti) {
    int n = wave * 32 + ti * 16 + ln;
#pragma unroll
    for (int r = 0; r < 8; ++r) {
      int m = mb + r;
      cF[m * H_ + n] = creg[ti][r];
      hF[m * H_ + n] = bf2f(sh_h[m * H_ + n]);
    }
  }
}

// -------------------------------------------------- state reduction + cov init
__global__ __launch_bounds__(256) void reduce_state_kernel(
    const float* __restrict__ hFf, const float* __restrict__ hFb,
    const float* __restrict__ cFf, const float* __restrict__ cFb,
    const float* __restrict__ red_hW, const float* __restrict__ red_hb,
    const float* __restrict__ red_cW, const float* __restrict__ red_cb,
    float* __restrict__ h_st, float* __restrict__ c_st, float* __restrict__ cov_st) {
  int idx = blockIdx.x * 256 + threadIdx.x;
  if (idx < B_ * H_) {
    int b = idx >> 8, n = idx & (H_ - 1);
    const float* wh = red_hW + (size_t)n * H2_;
    const float* wc = red_cW + (size_t)n * H2_;
    float ah = 0.f, ac = 0.f;
    for (int k = 0; k < H_; ++k) {
      ah += hFf[b * H_ + k] * wh[k];
      ac += cFf[b * H_ + k] * wc[k];
    }
    for (int k = 0; k < H_; ++k) {
      ah += hFb[b * H_ + k] * wh[H_ + k];
      ac += cFb[b * H_ + k] * wc[H_ + k];
    }
    h_st[idx] = fmaxf(ah + red_hb[n], 0.f);
    c_st[idx] = fmaxf(ac + red_cb[n], 0.f);
  }
  for (int i = idx; i < B_ * L_; i += gridDim.x * 256) cov_st[i] = 0.f;
}

__global__ void concat_hidden_kernel(const float* __restrict__ hs_f,
                                     const float* __restrict__ hs_b,
                                     float* __restrict__ enc_hid_f,
                                     unsigned short* __restrict__ enc_hid_bf) {
  int idx = blockIdx.x * blockDim.x + threadIdx.x;   // [B*L, 2H]
  if (idx >= B_ * L_ * H2_) return;
  int j = idx & (H2_ - 1);
  int bl = idx >> 9;
  int b = bl / L_, l = bl - b * L_;
  float v = (j < H_) ? hs_f[((size_t)l * B_ + b) * H_ + j]
                     : hs_b[((size_t)l * B_ + b) * H_ + (j - H_)];
  enc_hid_f[idx] = v;
  enc_hid_bf[idx] = f2bf(v);
}

// --------------------------------------------------- enc_feat = enc_hidden @ Wp^T
__global__ __launch_bounds__(256) void enc_feat_kernel(
    const unsigned short* __restrict__ enc_hid_bf,  // [B*L, 2H]
    const unsigned short* __restrict__ eproj_bf,    // [2H, 2H]
    float* __restrict__ enc_feat) {                 // [B*L, 2H]
  int lane = threadIdx.x & 31, wave = threadIdx.x >> 5;
  int ln = lane & 15, mb = (lane >> 4) << 3;
  int m0 = blockIdx.x * 16;
  const unsigned short* abase = enc_hid_bf + (size_t)m0 * H2_;
  if (blockIdx.x + 1 < gridDim.x)   // stream next A block (global_prefetch_b8)
    __builtin_prefetch(abase + (size_t)16 * H2_ + (size_t)(threadIdx.x & 31) * 16, 0, 1);
#pragma unroll
  for (int j = 0; j < 4; ++j) {
    int n0 = (wave * 4 + j) * 16;
    v8f acc = zero8();
    for (int k0 = 0; k0 < H2_; k0 += 32)
      acc = WMMA_BF16(load_a(abase, H2_, k0), load_b(eproj_bf, H2_, n0, k0), acc);
    int n = n0 + ln;
#pragma unroll
    for (int r = 0; r < 8; ++r)
      enc_feat[((size_t)m0 + mb + r) * H2_ + n] = acc[r];
  }
}

// ------------------------------------------------ fused decoder step (1 WGP)
__global__ __launch_bounds__(256) void dec_step_kernel(
    int t,
    const unsigned short* __restrict__ dec_emb_bf,  // [T,B,E]
    const float* __restrict__ dec_emb_f,            // [T,B,E]
    const unsigned short* __restrict__ Wih,         // [4H,E]
    const unsigned short* __restrict__ Whh,         // [4H,H]
    const float* __restrict__ bias,                 // [4H]
    const unsigned short* __restrict__ dprojW_bf,   // [2H,H]
    const float* __restrict__ dprojb,               // [2H]
    const float* __restrict__ enc_feat,             // [B*L,2H]
    const float* __restrict__ enc_hidden,           // [B*L,2H]
    const float* __restrict__ attn_v, const float* __restrict__ w_c,
    const unsigned short* __restrict__ vW_bf,       // [H,3H]
    const float* __restrict__ vb,
    const float* __restrict__ w_h, const float* __restrict__ w_s,
    const float* __restrict__ w_x, const float* __restrict__ b_x,
    float* __restrict__ h_st, float* __restrict__ c_st,
    float* __restrict__ cov_st, float* __restrict__ ad_st,
    unsigned short* __restrict__ out_bf, float* __restrict__ pgen_st,
    float* __restrict__ d_attn, float* __restrict__ d_cov) {
  __shared__ __align__(16) unsigned short sh_h[B_ * H_];     // 8 KB  h (bf16)
  __shared__ __align__(16) unsigned short sh_scr[B_ * H2_];  // 16 KB dfeat -> ctx (bf16)
  __shared__ float sh_s[B_ * L_];                            // 25.6 KB scores -> attn
  __shared__ float sh_pg[B_];
  const int tid = threadIdx.x, lane = tid & 31, wave = tid >> 5;
  const int ln = lane & 15, mb = (lane >> 4) << 3;

  // phase 0: state in
  for (int i = tid; i < B_ * H_; i += 256) sh_h[i] = f2bf(h_st[i]);
  if (tid < B_) sh_pg[tid] = 0.f;
  __syncthreads();

  // phase 1: LSTM gates via WMMA (wave w owns columns [w*32, w*32+32) of all 4 gates)
  {
    v8f acc[8];
#pragma unroll
    for (int i = 0; i < 8; ++i) acc[i] = zero8();
    const unsigned short* xrow = dec_emb_bf + (size_t)t * (B_ * E_);
#pragma unroll
    for (int k0 = 0; k0 < E_; k0 += 32) {
      v16bf a = load_a(xrow, E_, k0);
#pragma unroll
      for (int ti = 0; ti < 2; ++ti)
#pragma unroll
        for (int g = 0; g < 4; ++g)
          acc[ti * 4 + g] = WMMA_BF16(a, load_b(Wih, E_, g * H_ + wave * 32 + ti * 16, k0),
                                      acc[ti * 4 + g]);
    }
#pragma unroll 2
    for (int k0 = 0; k0 < H_; k0 += 32) {
      v16bf a = load_a(sh_h, H_, k0);
#pragma unroll
      for (int ti = 0; ti < 2; ++ti)
#pragma unroll
        for (int g = 0; g < 4; ++g)
          acc[ti * 4 + g] = WMMA_BF16(a, load_b(Whh, H_, g * H_ + wave * 32 + ti * 16, k0),
                                      acc[ti * 4 + g]);
    }
    __syncthreads();
#pragma unroll
    for (int ti = 0; ti < 2; ++ti) {
      int n = wave * 32 + ti * 16 + ln;
      float bi = bias[0 * H_ + n], bf = bias[1 * H_ + n];
      float bg = bias[2 * H_ + n], bo = bias[3 * H_ + n];
#pragma unroll
      for (int r = 0; r < 8; ++r) {
        int m = mb + r;
        float cp = c_st[m * H_ + n];
        float ig = sigm(acc[ti * 4 + 0][r] + bi);
        float fg = sigm(acc[ti * 4 + 1][r] + bf);
        float gg = tanhf(acc[ti * 4 + 2][r] + bg);
        float og = sigm(acc[ti * 4 + 3][r] + bo);
        float c = fg * cp + ig * gg;
        float h = og * tanhf(c);
        c_st[m * H_ + n] = c;
        h_st[m * H_ + n] = h;
        sh_h[m * H_ + n] = f2bf(h);
      }
    }
    __syncthreads();
  }

  // phase 2: dfeat = h @ dec_projW^T + dec_projb  -> sh_scr (bf16)
#pragma unroll
  for (int j = 0; j < 4; ++j) {
    int n0 = (wave * 4 + j) * 16;
    v8f acc = zero8();
#pragma unroll
    for (int k0 = 0; k0 < H_; k0 += 32)
      acc = WMMA_BF16(load_a(sh_h, H_, k0), load_b(dprojW_bf, H_, n0, k0), acc);
    int n = n0 + ln;
    float bb = dprojb[n];
#pragma unroll
    for (int r = 0; r < 8; ++r)
      sh_scr[(mb + r) * H2_ + n] = f2bf(acc[r] + bb);
  }
  __syncthreads();

  // phase 3: attention scores (enc_pad_mask is all-false in this problem)
  for (int p = wave; p < B_ * L_; p += 8) {
    int b = p / L_;
    const float* ef = enc_feat + (size_t)p * H2_;
    float cov = cov_st[p];
    float acc = 0.f;
    for (int j = lane; j < H2_; j += 32) {
      float xv = ef[j] + bf2f(sh_scr[b * H2_ + j]) + cov * w_c[j];
      acc += tanhf(xv) * attn_v[j];
    }
#pragma unroll
    for (int off = 16; off; off >>= 1) acc += __shfl_xor(acc, off);
    if (lane == 0) sh_s[p] = acc;
  }
  __syncthreads();

  // phase 4: softmax over L per batch row (wave-parallel, wave32 reductions)
  for (int rb = wave; rb < B_; rb += 8) {
    float mx = -1e30f;
    for (int l = lane; l < L_; l += 32) mx = fmaxf(mx, sh_s[rb * L_ + l]);
#pragma unroll
    for (int off = 16; off; off >>= 1) mx = fmaxf(mx, __shfl_xor(mx, off));
    float sum = 0.f;
    for (int l = lane; l < L_; l += 32) sum += __expf(sh_s[rb * L_ + l] - mx);
#pragma unroll
    for (int off = 16; off; off >>= 1) sum += __shfl_xor(sum, off);
    float inv = 1.0f / sum;
    for (int l = lane; l < L_; l += 32)
      sh_s[rb * L_ + l] = __expf(sh_s[rb * L_ + l] - mx) * inv;
  }
  __syncthreads();

  // phase 5: context (overwrites sh_scr with ctx bf16) + p_gen partial dots
  for (int idx = tid; idx < B_ * H2_; idx += 256) {
    int b = idx >> 9, j = idx & (H2_ - 1);
    const float* eh = enc_hidden + (size_t)b * L_ * H2_ + j;
    float acc = 0.f;
    for (int l = 0; l < L_; ++l) acc += sh_s[b * L_ + l] * eh[(size_t)l * H2_];
    sh_scr[idx] = f2bf(acc);
    atomicAdd(&sh_pg[b], acc * w_h[j]);          // ds_add_f32
  }
  for (int idx = tid; idx < B_ * H_; idx += 256) {
    int b = idx >> 8, n = idx & (H_ - 1);
    atomicAdd(&sh_pg[b], bf2f(sh_h[idx]) * w_s[n]);
  }
  for (int idx = tid; idx < B_ * E_; idx += 256) {
    int b = idx >> 7, e = idx & (E_ - 1);
    atomicAdd(&sh_pg[b], dec_emb_f[((size_t)t * B_ + b) * E_ + e] * w_x[e]);
  }
  __syncthreads();

  // phase 6: p_gen
  if (tid < B_) {
    float pg = sigm(sh_pg[tid] + b_x[0]);
    sh_pg[tid] = pg;
    pgen_st[tid] = pg;
  }
  __syncthreads();

  // phase 7: coverage update + attn/cov outputs (attns output == ad)
  for (int i = tid; i < B_ * L_; i += 256) {
    int b = i / L_;
    float a = sh_s[i];
    float cn = cov_st[i] + a;
    cov_st[i] = cn;
    float ad = (1.f - sh_pg[b]) * a;
    ad_st[i] = ad;
    d_attn[(size_t)i * T_ + t] = ad;
    d_cov[(size_t)i * T_ + t] = cn;
  }

  // phase 8: out = [h|ctx] @ vW^T + vb (A split across sh_h and sh_scr)
#pragma unroll
  for (int j = 0; j < 2; ++j) {
    int n0 = (wave * 2 + j) * 16;
    v8f acc = zero8();
#pragma unroll
    for (int k0 = 0; k0 < H_; k0 += 32)
      acc = WMMA_BF16(load_a(sh_h, H_, k0), load_b(vW_bf, H3_, n0, k0), acc);
#pragma unroll 2
    for (int k0 = 0; k0 < H2_; k0 += 32)
      acc = WMMA_BF16(load_a(sh_scr, H2_, k0), load_b(vW_bf, H3_, n0, H_ + k0), acc);
    int n = n0 + ln;
    float bb = vb[n];
#pragma unroll
    for (int r = 0; r < 8; ++r)
      out_bf[(mb + r) * H_ + n] = f2bf(acc[r] + bb);
  }
}

// ------------------- vocab GEMM [16,256]x[256,V] with TDM double buffering ---
// 4 waves; chunk = 64 rows of voutW (32 KB bf16) DMA'd into LDS by the Tensor
// Data Mover while WMMA consumes the other buffer. Wave 0 issues/waits on
// TENSORcnt; the workgroup barrier publishes the LDS tile to all waves.
__global__ __launch_bounds__(128) void vocab_kernel_tdm(
    const unsigned short* __restrict__ out_bf,   // [B,H]
    const unsigned short* __restrict__ voutW_bf, // [V,H]
    const float* __restrict__ voutb,
    float* __restrict__ logits) {                // [B,V]
  __shared__ __align__(16) unsigned short sbuf[2][VCHUNK_ROWS * H_];  // 64 KB
  const int lane = threadIdx.x & 31, wave = threadIdx.x >> 5;
  const int ln = lane & 15, mb = (lane >> 4) << 3;
  const int c0 = blockIdx.x * VNITER;

#if HAVE_TDM
  if (wave == 0 && c0 < VCHUNKS)
    tdm_issue(voutW_bf + (size_t)c0 * VCHUNK_ROWS * H_, 0u);
#endif

  for (int i = 0; i < VNITER; ++i) {
    const int c = c0 + i;
    const bool have_cur = (c < VCHUNKS);
#if HAVE_TDM
    const bool have_next = (i + 1 < VNITER) && (c + 1 < VCHUNKS);
    if (wave == 0) {
      if (have_next)
        tdm_issue(voutW_bf + (size_t)(c + 1) * VCHUNK_ROWS * H_,
                  (unsigned)(((i + 1) & 1) * (VCHUNK_ROWS * H_ * 2)));
      if (have_cur) {
        if (have_next) __builtin_amdgcn_s_wait_tensorcnt(1);  // cur complete (in-order)
        else           __builtin_amdgcn_s_wait_tensorcnt(0);
      }
    }
#else
    if (have_cur) {
      const size_t gbase = (size_t)c * VCHUNK_ROWS * H_;
      for (int idx = threadIdx.x; idx < VCHUNK_ROWS * H_; idx += 128) {
        size_t g = gbase + idx;
        sbuf[i & 1][idx] = (g < (size_t)V_ * H_) ? voutW_bf[g] : (unsigned short)0;
      }
      __builtin_prefetch(voutW_bf + gbase + VCHUNK_ROWS * H_ + (size_t)threadIdx.x * 16, 0, 1);
    }
#endif
    __syncthreads();

    const int tile = c * 4 + wave;             // 4 N-tiles per chunk
    if (have_cur && tile * 16 < V_) {
      const unsigned short* bbase = &sbuf[i & 1][0];
      v8f acc = zero8();
#pragma unroll
      for (int k0 = 0; k0 < H_; k0 += 32)
        acc = WMMA_BF16(load_a(out_bf, H_, k0), load_b(bbase, H_, wave * 16, k0), acc);
      int n = tile * 16 + ln;
      float bb = voutb[n];
#pragma unroll
      for (int r = 0; r < 8; ++r)
        logits[(size_t)(mb + r) * V_ + n] = acc[r] + bb;
    }
    __syncthreads();   // tile consumed before the TDM for i+2 may overwrite it
  }
}

__global__ __launch_bounds__(256) void softmax_stats_kernel(
    const float* __restrict__ logits, float* __restrict__ rmax, float* __restrict__ rsum) {
  __shared__ float red[256];
  int b = blockIdx.x, tid = threadIdx.x;
  const float* row = logits + (size_t)b * V_;
  float mx = -1e30f;
  for (int v = tid; v < V_; v += 256) mx = fmaxf(mx, row[v]);
  red[tid] = mx; __syncthreads();
  for (int s = 128; s; s >>= 1) { if (tid < s) red[tid] = fmaxf(red[tid], red[tid + s]); __syncthreads(); }
  mx = red[0]; __syncthreads();
  float sum = 0.f;
  for (int v = tid; v < V_; v += 256) sum += __expf(row[v] - mx);
  red[tid] = sum; __syncthreads();
  for (int s = 128; s; s >>= 1) { if (tid < s) red[tid] += red[tid + s]; __syncthreads(); }
  if (tid == 0) { rmax[b] = mx; rsum[b] = red[0]; }
}

__global__ void final_write_kernel(int t, const float* __restrict__ logits,
                                   const float* __restrict__ rmax,
                                   const float* __restrict__ rsum,
                                   const float* __restrict__ pgen,
                                   float* __restrict__ finals) {
  int idx = blockIdx.x * blockDim.x + threadIdx.x;   // [B, V+OOV]
  if (idx >= B_ * VEXT_) return;
  int b = idx / VEXT_, v = idx - b * VEXT_;
  float val = 0.f;
  if (v < V_) val = pgen[b] * __expf(logits[(size_t)b * V_ + v] - rmax[b]) / rsum[b];
  finals[(size_t)idx * T_ + t] = val;   // layout [B, V+OOV, T]
}

__global__ void scatter_kernel(int t, const int* __restrict__ enc_ext,
                               const float* __restrict__ ad_st,
                               float* __restrict__ finals) {
  int i = blockIdx.x * blockDim.x + threadIdx.x;   // B*L
  if (i >= B_ * L_) return;
  int b = i / L_;
  int v = enc_ext[i];
  atomicAdd(finals + ((size_t)(b * VEXT_ + v)) * T_ + t, ad_st[i]);
}

// =============================================================== host driver
extern "C" void kernel_launch(void* const* d_in, const int* in_sizes, int n_in,
                              void* d_out, int out_size, void* d_ws, size_t ws_size,
                              hipStream_t stream) {
  (void)in_sizes; (void)n_in; (void)out_size; (void)ws_size;
  const int*   enc_input     = (const int*)d_in[0];
  const int*   enc_input_ext = (const int*)d_in[1];
  // d_in[2] enc_pad_mask (all false), d_in[3] enc_len (full), d_in[5] max_oov_len: unused
  const int*   dec_input = (const int*)d_in[4];
  const float* emb    = (const float*)d_in[6];
  const float* eWihF  = (const float*)d_in[7];
  const float* eWhhF  = (const float*)d_in[8];
  const float* ebF    = (const float*)d_in[9];
  const float* eWihB  = (const float*)d_in[10];
  const float* eWhhB  = (const float*)d_in[11];
  const float* ebB    = (const float*)d_in[12];
  const float* red_hW = (const float*)d_in[13];
  const float* red_hb = (const float*)d_in[14];
  const float* red_cW = (const float*)d_in[15];
  const float* red_cb = (const float*)d_in[16];
  const float* dWih   = (const float*)d_in[17];
  const float* dWhh   = (const float*)d_in[18];
  const float* db     = (const float*)d_in[19];
  const float* attn_v = (const float*)d_in[20];
  const float* eprojW = (const float*)d_in[21];
  const float* dprojW = (const float*)d_in[22];
  const float* dprojb = (const float*)d_in[23];
  const float* w_c    = (const float*)d_in[24];
  const float* vW     = (const float*)d_in[25];
  const float* vb     = (const float*)d_in[26];
  const float* voutW  = (const float*)d_in[27];
  const float* voutb  = (const float*)d_in[28];
  const float* w_h    = (const float*)d_in[29];
  const float* w_s    = (const float*)d_in[30];
  const float* w_x    = (const float*)d_in[31];
  const float* b_x    = (const float*)d_in[32];

  char* ws = (char*)d_ws;
  size_t off = 0;
  auto alloc = [&](size_t bytes) -> void* {
    void* p = ws + off;
    off += (bytes + 255) & ~(size_t)255;
    return p;
  };
  // bf16 buffers
  unsigned short* voutW_bf  = (unsigned short*)alloc((size_t)V_ * H_ * 2);
  unsigned short* eWihF_bf  = (unsigned short*)alloc((size_t)G4H_ * E_ * 2);
  unsigned short* eWhhF_bf  = (unsigned short*)alloc((size_t)G4H_ * H_ * 2);
  unsigned short* eWihB_bf  = (unsigned short*)alloc((size_t)G4H_ * E_ * 2);
  unsigned short* eWhhB_bf  = (unsigned short*)alloc((size_t)G4H_ * H_ * 2);
  unsigned short* dWih_bf   = (unsigned short*)alloc((size_t)G4H_ * E_ * 2);
  unsigned short* dWhh_bf   = (unsigned short*)alloc((size_t)G4H_ * H_ * 2);
  unsigned short* dprojW_bf = (unsigned short*)alloc((size_t)H2_ * H_ * 2);
  unsigned short* eprojW_bf = (unsigned short*)alloc((size_t)H2_ * H2_ * 2);
  unsigned short* vW_bf     = (unsigned short*)alloc((size_t)H_ * H3_ * 2);
  unsigned short* enc_emb_bf = (unsigned short*)alloc((size_t)L_ * B_ * E_ * 2);
  unsigned short* dec_emb_bf = (unsigned short*)alloc((size_t)T_ * B_ * E_ * 2);
  unsigned short* enc_hid_bf = (unsigned short*)alloc((size_t)B_ * L_ * H2_ * 2);
  unsigned short* out_bf     = (unsigned short*)alloc((size_t)B_ * H_ * 2);
  // f32 buffers
  float* dec_emb_f = (float*)alloc((size_t)T_ * B_ * E_ * 4);
  float* hs_f      = (float*)alloc((size_t)L_ * B_ * H_ * 4);
  float* hs_b      = (float*)alloc((size_t)L_ * B_ * H_ * 4);
  float* hFf = (float*)alloc((size_t)B_ * H_ * 4);
  float* hFb = (float*)alloc((size_t)B_ * H_ * 4);
  float* cFf = (float*)alloc((size_t)B_ * H_ * 4);
  float* cFb = (float*)alloc((size_t)B_ * H_ * 4);
  float* enc_hid_f = (float*)alloc((size_t)B_ * L_ * H2_ * 4);
  float* enc_feat  = (float*)alloc((size_t)B_ * L_ * H2_ * 4);
  float* h_st   = (float*)alloc((size_t)B_ * H_ * 4);
  float* c_st   = (float*)alloc((size_t)B_ * H_ * 4);
  float* cov_st = (float*)alloc((size_t)B_ * L_ * 4);
  float* ad_st  = (float*)alloc((size_t)B_ * L_ * 4);
  float* logits = (float*)alloc((size_t)B_ * V_ * 4);
  float* pgen_st = (float*)alloc(B_ * 4);
  float* rmax    = (float*)alloc(B_ * 4);
  float* rsum    = (float*)alloc(B_ * 4);

  auto cast = [&](const float* s, unsigned short* d, int n) {
    cast_bf16_kernel<<<(n + 255) / 256, 256, 0, stream>>>(s, d, n);
  };
  cast(voutW, voutW_bf, V_ * H_);
  cast(eWihF, eWihF_bf, G4H_ * E_);
  cast(eWhhF, eWhhF_bf, G4H_ * H_);
  cast(eWihB, eWihB_bf, G4H_ * E_);
  cast(eWhhB, eWhhB_bf, G4H_ * H_);
  cast(dWih, dWih_bf, G4H_ * E_);
  cast(dWhh, dWhh_bf, G4H_ * H_);
  cast(dprojW, dprojW_bf, H2_ * H_);
  cast(eprojW, eprojW_bf, H2_ * H2_);
  cast(vW, vW_bf, H_ * H3_);

  embed_enc_kernel<<<(L_ * B_ * E_ + 255) / 256, 256, 0, stream>>>(enc_input, emb, enc_emb_bf);
  embed_dec_kernel<<<(T_ * B_ * E_ + 255) / 256, 256, 0, stream>>>(dec_input, emb, dec_emb_bf,
                                                                   dec_emb_f);
  enc_lstm_kernel<<<1, 256, 0, stream>>>(enc_emb_bf, eWihF_bf, eWhhF_bf, ebF, hs_f, hFf, cFf, 0);
  enc_lstm_kernel<<<1, 256, 0, stream>>>(enc_emb_bf, eWihB_bf, eWhhB_bf, ebB, hs_b, hFb, cFb, 1);
  reduce_state_kernel<<<16, 256, 0, stream>>>(hFf, hFb, cFf, cFb, red_hW, red_hb, red_cW, red_cb,
                                              h_st, c_st, cov_st);
  concat_hidden_kernel<<<(B_ * L_ * H2_ + 255) / 256, 256, 0, stream>>>(hs_f, hs_b, enc_hid_f,
                                                                        enc_hid_bf);
  enc_feat_kernel<<<(B_ * L_) / 16, 256, 0, stream>>>(enc_hid_bf, eprojW_bf, enc_feat);

  float* d_finals = (float*)d_out;
  float* d_attn = d_finals + (size_t)B_ * VEXT_ * T_;
  float* d_cov  = d_attn + (size_t)B_ * L_ * T_;

  for (int t = 0; t < T_; ++t) {
    dec_step_kernel<<<1, 256, 0, stream>>>(
        t, dec_emb_bf, dec_emb_f, dWih_bf, dWhh_bf, db, dprojW_bf, dprojb, enc_feat, enc_hid_f,
        attn_v, w_c, vW_bf, vb, w_h, w_s, w_x, b_x, h_st, c_st, cov_st, ad_st, out_bf, pgen_st,
        d_attn, d_cov);
    vocab_kernel_tdm<<<VBLOCKS, 128, 0, stream>>>(out_bf, voutW_bf, voutb, logits);
    softmax_stats_kernel<<<B_, 256, 0, stream>>>(logits, rmax, rsum);
    final_write_kernel<<<(B_ * VEXT_ + 255) / 256, 256, 0, stream>>>(t, logits, rmax, rsum,
                                                                     pgen_st, d_finals);
    scatter_kernel<<<(B_ * L_ + 255) / 256, 256, 0, stream>>>(t, enc_input_ext, ad_st, d_finals);
  }
}
